// BLoraLinear_46471546143180
// MI455X (gfx1250) — matrix operand order repaired and checked by reading, not compiled, not run on hardware
//
#include <hip/hip_runtime.h>

// ---------------------------------------------------------------------------
// BLoRA ragged linear for MI455X (gfx1250, wave32, WMMA)
//   out = x @ W^T + b  (+ per-segment LoRA:  (x @ A_s) @ B_s summed over M)
// Path: fp32 HBM -> bf16 LDS tiles -> v_wmma_f32_16x16x32_bf16 (fp32 accum)
// K-stage = 64 (2 WMMA sub-steps) -> 16 WMMAs per block barrier per wave.
// ---------------------------------------------------------------------------

typedef __attribute__((ext_vector_type(16))) __bf16 v16bf;
typedef __attribute__((ext_vector_type(8)))  __bf16 v8bf;
typedef __attribute__((ext_vector_type(4)))  __bf16 v4bf;
typedef __attribute__((ext_vector_type(8)))  float  v8f;
typedef __attribute__((ext_vector_type(4)))  float  f4;

#define T_TOK 8192
#define D_IN  4096
#define D_OUT 4096
#define RNK   16
#define NMOD  2
#define NSEG  8

#define BT 128      // token tile (base GEMM)
#define BO 128      // output tile (base GEMM)
#define BK 64       // K elements staged per barrier (2 x 32-wide WMMA steps)
#define WK 32       // WMMA K for bf16
#define LSTR 72     // LDS row stride in bf16 elems (144B: 16B-aligned, bank-spread)

static __device__ __forceinline__ v8f wmma_bf16(v16bf a, v16bf b, v8f c) {
  // 8 args: (neg_a, A, neg_b, B, c_mod, C, reuse_a, reuse_b)
  return __builtin_amdgcn_wmma_f32_16x16x32_bf16(false, a, false, b, (short)0, c,
                                                 false, false);
}

static __device__ __forceinline__ v8f zero8() {
  v8f z;
#pragma unroll
  for (int i = 0; i < 8; ++i) z[i] = 0.0f;
  return z;
}

// ---------------------------------------------------------------------------
// Kernel 1: base GEMM  out[t,o] = sum_d x[t,d] * W[o,d] + b[o]
// 256 threads = 8 waves; block tile 128x128; wave tile 64x32 (4x2 WMMA tiles)
// ---------------------------------------------------------------------------
__global__ void __launch_bounds__(256)
blora_base_gemm(const float* __restrict__ x, const float* __restrict__ W,
                const float* __restrict__ bias, float* __restrict__ out)
{
  __shared__ __bf16 xs[2][BT * LSTR];
  __shared__ __bf16 wsh[2][BO * LSTR];

  const int tid  = threadIdx.x;
  const int lane = tid & 31;
  const int wave = tid >> 5;
  const int tb = blockIdx.y * BT;
  const int ob = blockIdx.x * BO;
  const int wr = wave >> 2;          // 0..1 -> 64-token slab
  const int wc = wave & 3;           // 0..3 -> 32-output slab

  v8f acc[4][2];
#pragma unroll
  for (int i = 0; i < 4; ++i)
#pragma unroll
    for (int j = 0; j < 2; ++j) acc[i][j] = zero8();

  // global -> LDS staging (with fp32->bf16 convert); 2 threads per row,
  // each thread covers 32 of the 64 staged K columns (8 float4 per matrix).
  const int lrow  = tid >> 1;          // 0..127
  const int lcol0 = (tid & 1) * 32;    // 0 or 32

  auto stage = [&](int buf, int k0) {
    const float* xp = x + (size_t)(tb + lrow) * D_IN + k0 + lcol0;
    const float* wp = W + (size_t)(ob + lrow) * D_IN + k0 + lcol0;
#pragma unroll
    for (int i = 0; i < 8; ++i) {
      f4 xv = *(const f4*)(xp + 4 * i);
      f4 wv = *(const f4*)(wp + 4 * i);
      v4bf xc, wc4;
#pragma unroll
      for (int e = 0; e < 4; ++e) {
        xc[e]  = (__bf16)xv[e];
        wc4[e] = (__bf16)wv[e];
      }
      *(v4bf*)&xs[buf][lrow * LSTR + lcol0 + 4 * i]  = xc;
      *(v4bf*)&wsh[buf][lrow * LSTR + lcol0 + 4 * i] = wc4;
    }
    if (k0 + 2 * BK < D_IN) {          // look-ahead prefetch (global_prefetch_b8)
      __builtin_prefetch(xp + 2 * BK, 0, 0);
      __builtin_prefetch(wp + 2 * BK, 0, 0);
    }
  };

  stage(0, 0);
  __syncthreads();

  const int l16   = lane & 15;
  const int ahalf = lane >> 4;         // 0/1 half-wave (K-half select)

  for (int k0 = 0; k0 < D_IN; k0 += BK) {
    const int buf = (k0 / BK) & 1;
    if (k0 + BK < D_IN) stage(buf ^ 1, k0 + BK);

#pragma unroll
    for (int kk = 0; kk < BK; kk += WK) {
      // A-frag (16x32 bf16): lane = row l16; K = kk + ahalf*16..+15 (contig)
      v16bf af[4];
#pragma unroll
      for (int rt = 0; rt < 4; ++rt) {
        const __bf16* p =
            &xs[buf][(wr * 64 + rt * 16 + l16) * LSTR + kk + ahalf * 16];
        v8bf lo = *(const v8bf*)p;          // 16B
        v8bf hi = *(const v8bf*)(p + 8);    // 16B at +16B
        af[rt] = __builtin_shufflevector(lo, hi,
                   0, 1, 2, 3, 4, 5, 6, 7, 8, 9, 10, 11, 12, 13, 14, 15);
      }
      // B-frag (32x16 bf16): lane = column l16; K = kk + ahalf*16..+15 (contig)
      v16bf bfr[2];
#pragma unroll
      for (int ct = 0; ct < 2; ++ct) {
        const __bf16* p =
            &wsh[buf][(wc * 32 + ct * 16 + l16) * LSTR + kk + ahalf * 16];
        bfr[ct] = *(const v16bf*)p;         // 32B
      }
#pragma unroll
      for (int rt = 0; rt < 4; ++rt)
#pragma unroll
        for (int ct = 0; ct < 2; ++ct)
          acc[rt][ct] = wmma_bf16(af[rt], bfr[ct], acc[rt][ct]);
    }
    __syncthreads();
  }

  // Epilogue: C/D layout -> lane l16 = column, VGPR j = row j (+8 for hi half)
#pragma unroll
  for (int rt = 0; rt < 4; ++rt) {
#pragma unroll
    for (int ct = 0; ct < 2; ++ct) {
      const int col  = ob + wc * 32 + ct * 16 + l16;
      const float bv = bias[col];
      float* op = out + (size_t)(tb + wr * 64 + rt * 16 + ahalf * 8) * D_OUT + col;
#pragma unroll
      for (int j = 0; j < 8; ++j) op[(size_t)j * D_OUT] = acc[rt][ct][j] + bv;
    }
  }
}

// ---------------------------------------------------------------------------
// Kernel 2: LoRA down-projection  U[t, m*16+r] = sum_d x[t,d] * A[m,seg(t),d,r]
// One wave per 16-token tile; exact raggedness via per-row zero-masking over
// the (wave-uniform) set of segments overlapping the tile.
// ---------------------------------------------------------------------------
__global__ void __launch_bounds__(256)
blora_lora_down(const float* __restrict__ x, const float* __restrict__ A,
                const int* __restrict__ cu, float* __restrict__ U)
{
  const int lane = threadIdx.x & 31;
  const int wave = threadIdx.x >> 5;
  const int t0   = (blockIdx.x * 8 + wave) * 16;

  int cs[NSEG + 1];
#pragma unroll
  for (int i = 0; i <= NSEG; ++i) cs[i] = cu[i];

  const int l16   = lane & 15;
  const int ahalf = lane >> 4;
  const int arow  = t0 + l16;          // token row held by this lane (A-frag)

  v8f acc0 = zero8(), acc1 = zero8();

  for (int s = 0; s < NSEG; ++s) {
    if (cs[s] >= t0 + 16 || cs[s + 1] <= t0) continue;   // wave-uniform branch
    const float msk = (arow >= cs[s] && arow < cs[s + 1]) ? 1.0f : 0.0f;
    const float* a0 = A + (size_t)(0 * NSEG + s) * D_IN * RNK;
    const float* a1 = A + (size_t)(1 * NSEG + s) * D_IN * RNK;

    for (int k0 = 0; k0 < D_IN; k0 += WK) {
      // A-frag from x (row-masked), bf16-converted
      const float* xp = x + (size_t)arow * D_IN + k0 + ahalf * 16;
      f4 xa = *(const f4*)xp;
      f4 xb = *(const f4*)(xp + 4);
      f4 xc = *(const f4*)(xp + 8);
      f4 xd = *(const f4*)(xp + 12);
      v16bf af;
#pragma unroll
      for (int e = 0; e < 4; ++e) {
        af[e]      = (__bf16)(xa[e] * msk);
        af[e + 4]  = (__bf16)(xb[e] * msk);
        af[e + 8]  = (__bf16)(xc[e] * msk);
        af[e + 12] = (__bf16)(xd[e] * msk);
      }
      // B-frags: lane = rank column r, K = ahalf*16 + j (stride RNK in memory)
      v16bf b0, b1;
      const int kb = k0 + ahalf * 16;
#pragma unroll
      for (int j = 0; j < 16; ++j) {
        b0[j] = (__bf16)a0[(size_t)(kb + j) * RNK + l16];
        b1[j] = (__bf16)a1[(size_t)(kb + j) * RNK + l16];
      }
      acc0 = wmma_bf16(af, b0, acc0);
      acc1 = wmma_bf16(af, b1, acc1);
    }
  }

  // U rows: row = t0 + ahalf*8 + j, cols: m*16 + l16
  float* up = U + (size_t)(t0 + ahalf * 8) * (NMOD * RNK) + l16;
#pragma unroll
  for (int j = 0; j < 8; ++j) {
    up[(size_t)j * (NMOD * RNK) + 0]   = acc0[j];
    up[(size_t)j * (NMOD * RNK) + 16]  = acc1[j];
  }
}

// ---------------------------------------------------------------------------
// Kernel 3: LoRA up-projection  out[t,o] += sum_{m,r} U[t,m*16+r]*B[m,seg(t),r,o]
// Single K=32 WMMA per 16x16 tile per overlapping segment; the output tile is
// loaded straight into the C operand and stored back (read-modify-write).
// ---------------------------------------------------------------------------
__global__ void __launch_bounds__(256)
blora_lora_up(const float* __restrict__ U, const float* __restrict__ B,
              const int* __restrict__ cu, float* __restrict__ out)
{
  const int lane = threadIdx.x & 31;
  const int wave = threadIdx.x >> 5;
  const int t0   = blockIdx.y * 16;
  const int l16  = lane & 15;
  const int half = lane >> 4;
  const int ocol = blockIdx.x * 128 + wave * 16 + l16;

  int cs[NSEG + 1];
#pragma unroll
  for (int i = 0; i <= NSEG; ++i) cs[i] = cu[i];

  float* op = out + (size_t)(t0 + half * 8) * D_OUT + ocol;
  v8f acc;
#pragma unroll
  for (int j = 0; j < 8; ++j) acc[j] = op[(size_t)j * D_OUT];

  const int arow = t0 + l16;

  for (int s = 0; s < NSEG; ++s) {
    if (cs[s] >= t0 + 16 || cs[s + 1] <= t0) continue;   // wave-uniform branch
    const float msk = (arow >= cs[s] && arow < cs[s + 1]) ? 1.0f : 0.0f;

    // A-frag from U[arow][k], combined K axis k = m*16 + r (32 wide)
    const float* up = U + (size_t)arow * (NMOD * RNK) + half * 16;
    f4 ua = *(const f4*)up;
    f4 ub = *(const f4*)(up + 4);
    f4 uc = *(const f4*)(up + 8);
    f4 ud = *(const f4*)(up + 12);
    v16bf af;
#pragma unroll
    for (int e = 0; e < 4; ++e) {
      af[e]      = (__bf16)(ua[e] * msk);
      af[e + 4]  = (__bf16)(ub[e] * msk);
      af[e + 8]  = (__bf16)(uc[e] * msk);
      af[e + 12] = (__bf16)(ud[e] * msk);
    }
    // B-frag: lane = column o; k = half*16 + j -> m = half, r = j
    v16bf bfr;
    const float* bp = B + ((size_t)half * NSEG + s) * RNK * D_OUT + ocol;
#pragma unroll
    for (int j = 0; j < 16; ++j) bfr[j] = (__bf16)bp[(size_t)j * D_OUT];

    acc = wmma_bf16(af, bfr, acc);
  }

#pragma unroll
  for (int j = 0; j < 8; ++j) op[(size_t)j * D_OUT] = acc[j];
}

// ---------------------------------------------------------------------------
extern "C" void kernel_launch(void* const* d_in, const int* in_sizes, int n_in,
                              void* d_out, int out_size, void* d_ws, size_t ws_size,
                              hipStream_t stream) {
  (void)in_sizes; (void)n_in; (void)out_size; (void)ws_size;
  const float* x  = (const float*)d_in[0];
  const float* W  = (const float*)d_in[1];
  const float* b  = (const float*)d_in[2];
  const float* lA = (const float*)d_in[3];   // [M, S, D_IN, R]
  const float* lB = (const float*)d_in[4];   // [M, S, R, D_OUT]
  const int*   cu = (const int*)d_in[5];     // [S+1]
  float* out = (float*)d_out;
  float* U   = (float*)d_ws;                 // [T, M*R] fp32 scratch (1 MB)

  blora_base_gemm<<<dim3(D_OUT / BO, T_TOK / BT), 256, 0, stream>>>(x, W, b, out);
  blora_lora_down<<<dim3(T_TOK / 128), 256, 0, stream>>>(x, lA, cu, U);
  blora_lora_up<<<dim3(D_OUT / 128, T_TOK / 16), 256, 0, stream>>>(U, lB, cu, out);
}
